// HGAT_23837068493064
// MI455X (gfx1250) — matrix operand order repaired and checked by reading, not compiled, not run on hardware
//
#include <hip/hip_runtime.h>
#include <hip/hip_bf16.h>
#include <math.h>

typedef __bf16 bf16_t;
typedef __attribute__((ext_vector_type(16))) __bf16 v16bf;
typedef __attribute__((ext_vector_type(8)))  float  v8f;

#define NWAL 200000
#define NTOK 100000
#define DW   128
#define DT   64
#define HID  64
#define HC   128
#define NEDG 500000
#define NEG_SLOPE 0.2f

// ---------------------------------------------------------------------------
// Pack W[K,N] (f32 row-major) into bf16 WMMA B-fragment order:
// tile (kt,nt) covers K=[kt*32,kt*32+32) x N=[nt*16,nt*16+16).
// Within a tile: lane = col + 16*(kk>=16), elem = kk&15  (contiguous k = 16*half + elem, same order as the A load).
// Tile storage: [(kt*tilesN + nt)*32 + lane]*16 + elem  (1KB per tile).
// ---------------------------------------------------------------------------
__global__ void pack_w(const float* __restrict__ W, bf16_t* __restrict__ out,
                       int K, int N) {
    int i = blockIdx.x * blockDim.x + threadIdx.x;
    if (i >= K * N) return;
    int k = i / N, n = i - k * N;
    int kt = k >> 5, kk = k & 31;
    int nt = n >> 4, c  = n & 15;
    int lane = c + ((kk >> 4) << 4);
    int e    = kk & 15;
    int tilesN = N >> 4;
    out[(((size_t)(kt * tilesN + nt)) * 32 + lane) * 16 + e] = (bf16_t)W[i];
}

// ---------------------------------------------------------------------------
// WMMA GEMM: Y[M,N] = X[M,K] (f32, ld=K) * W (pre-packed bf16) [+ bias]
// One wave computes a 16x64 strip: 4 accumulators share one A fragment,
// 4 back-to-back v_wmma_f32_16x16x32_bf16 per 32-wide K step.
// ---------------------------------------------------------------------------
__global__ void gemm_wmma_bf16(const float* __restrict__ X, int M, int K,
                               const bf16_t* __restrict__ Wp, int N,
                               float* __restrict__ Y,
                               const float* __restrict__ bias) {
    const int lane = threadIdx.x & 31;
    const int wave = threadIdx.x >> 5;
    const int groupsN = N >> 6;                 // 64-wide column groups
    const int tilesM  = (M + 15) >> 4;
    const int tilesN  = N >> 4;
    int tile = blockIdx.x * 8 + wave;           // wave-uniform
    if (tile >= tilesM * groupsN) return;       // uniform branch: EXEC all-1s
    const int tm  = tile / groupsN;
    const int g   = tile - tm * groupsN;
    const int m0  = tm << 4;
    const int nt0 = g << 2;                     // first of 4 N-tiles

    const int r    = lane & 15;
    const int half = lane >> 4;
    int arow = m0 + r; if (arow >= M) arow = M - 1;   // clamp; stores guarded
    const float* __restrict__ xrow = X + (size_t)arow * K;

    v8f acc[4] = {{}, {}, {}, {}};
    const int ktiles = K >> 5;
    for (int kt = 0; kt < ktiles; ++kt) {
        const int k = kt << 5;
        if (k + 32 < K) __builtin_prefetch(xrow + k + 32, 0, 1);
        // A fragment (16x32 bf16): half 0 -> K [0..15]; half 1 -> [16..31]
        v16bf a;
        const float* p0 = xrow + k + half * 16;
        const float* p1 = xrow + k + half * 16 + 8;
#pragma unroll
        for (int i = 0; i < 8; ++i) a[i] = (bf16_t)p0[i];
#pragma unroll
        for (int i = 0; i < 8; ++i) a[8 + i] = (bf16_t)p1[i];
        // B fragments: one 32B vector load per lane per tile, tiles contiguous
        const bf16_t* bp = Wp + (((size_t)kt * tilesN + nt0) * 32 + lane) * 16;
        v16bf b0 = *(const v16bf*)(bp);
        v16bf b1 = *(const v16bf*)(bp + 512);
        v16bf b2 = *(const v16bf*)(bp + 1024);
        v16bf b3 = *(const v16bf*)(bp + 1536);
        acc[0] = __builtin_amdgcn_wmma_f32_16x16x32_bf16(false, a, false, b0, (short)0, acc[0], false, false);
        acc[1] = __builtin_amdgcn_wmma_f32_16x16x32_bf16(false, a, false, b1, (short)0, acc[1], false, false);
        acc[2] = __builtin_amdgcn_wmma_f32_16x16x32_bf16(false, a, false, b2, (short)0, acc[2], false, false);
        acc[3] = __builtin_amdgcn_wmma_f32_16x16x32_bf16(false, a, false, b3, (short)0, acc[3], false, false);
    }
#pragma unroll
    for (int t = 0; t < 4; ++t) {
        const int col = ((nt0 + t) << 4) + r;
        const float bv = bias ? bias[col] : 0.0f;
#pragma unroll
        for (int j = 0; j < 8; ++j) {           // C: VGPR j -> row m0+j (+8 for half 1)
            int row = m0 + j + half * 8;
            if (row < M) Y[(size_t)row * N + col] = acc[t][j] + bv;
        }
    }
}

// ---------------------------------------------------------------------------
// fold[k,h] = sum_c W[k, h*64+c] * att[h,c]   (W: [K,HC], att: [2,HID])
// ---------------------------------------------------------------------------
__global__ void fold_kernel(const float* __restrict__ W, const float* __restrict__ att,
                            float* __restrict__ fold, int K) {
    int t = blockIdx.x * blockDim.x + threadIdx.x;
    if (t >= K * 2) return;
    int k = t >> 1, h = t & 1;
    const float* wrow = W + (size_t)k * HC + h * HID;
    const float* arow = att + h * HID;
    float s = 0.f;
    for (int c = 0; c < HID; ++c) s += wrow[c] * arow[c];
    fold[2 * k + h] = s;
}

// a[n,h] = x[n,:] . fold[:,h]
__global__ void scores_kernel(const float* __restrict__ X, int Nn, int K,
                              const float* __restrict__ fold, float* __restrict__ a) {
    int n = blockIdx.x * blockDim.x + threadIdx.x;
    if (n >= Nn) return;
    const float* x = X + (size_t)n * K;
    float s0 = 0.f, s1 = 0.f;
    for (int k = 0; k < K; ++k) { float xv = x[k]; s0 += xv * fold[2*k]; s1 += xv * fold[2*k+1]; }
    a[2*n] = s0; a[2*n+1] = s1;
}

__global__ void init_md(float* __restrict__ m, float* __restrict__ den, int n2) {
    int i = blockIdx.x * blockDim.x + threadIdx.x;
    if (i < n2) { m[i] = -INFINITY; den[i] = 0.f; }
}

__global__ void bias_init(float* __restrict__ out, size_t total,
                          const float* __restrict__ b1, const float* __restrict__ b2) {
    size_t i = (size_t)blockIdx.x * blockDim.x + threadIdx.x;
    if (i >= total) return;
    int c = (int)(i & (HC - 1));
    out[i] = b1[c] + (b2 ? b2[c] : 0.f);
}

// sign-aware float atomic max (correct for any mix of signs; init = -inf)
__device__ __forceinline__ void atomicMaxF(float* addr, float val) {
    if (val >= 0.f) atomicMax((int*)addr, __float_as_int(val));
    else            atomicMin((unsigned int*)addr, __float_as_uint(val));
}

__global__ void edge_pass1(const int* __restrict__ src, const int* __restrict__ dst,
                           const float* __restrict__ a_s, const float* __restrict__ a_d,
                           float* __restrict__ alpha, float* __restrict__ m, int E) {
    int e = blockIdx.x * blockDim.x + threadIdx.x;
    if (e >= E) return;
    int s = src[e], d = dst[e];
#pragma unroll
    for (int h = 0; h < 2; ++h) {
        float v = a_s[2*s+h] + a_d[2*d+h];
        v = v > 0.f ? v : NEG_SLOPE * v;
        alpha[2*e+h] = v;
        atomicMaxF(&m[2*d+h], v);
    }
}

__global__ void edge_pass2(const int* __restrict__ dst, float* __restrict__ alpha_ex,
                           const float* __restrict__ m, float* __restrict__ den, int E) {
    int e = blockIdx.x * blockDim.x + threadIdx.x;
    if (e >= E) return;
    int d = dst[e];
#pragma unroll
    for (int h = 0; h < 2; ++h) {
        float mm = m[2*d+h];
        if (!(mm > -1e37f)) mm = 0.f;             // jnp.where(isfinite(m), m, 0)
        float ex = __expf(alpha_ex[2*e+h] - mm);
        alpha_ex[2*e+h] = ex;
        atomicAdd(&den[2*d+h], ex);
    }
}

// one wave per edge: out[d, :] += (ex/(den+eps)) * hs[s, :]  (128 floats, 4/lane)
__global__ void edge_pass3(const int* __restrict__ src, const int* __restrict__ dst,
                           const float* __restrict__ ex, const float* __restrict__ den,
                           const float* __restrict__ hs, float* __restrict__ out, int E) {
    int e = blockIdx.x * 8 + (threadIdx.x >> 5);
    if (e >= E) return;
    int lane = threadIdx.x & 31;
    int s = src[e], d = dst[e];
    int h = lane >> 4;                             // cols [0,64) head 0, [64,128) head 1
    float w = ex[2*e+h] / (den[2*d+h] + 1e-16f);
    float4 v = reinterpret_cast<const float4*>(hs + (size_t)s * HC)[lane];
    float* o = out + (size_t)d * HC + lane * 4;
    atomicAdd(o + 0, v.x * w);
    atomicAdd(o + 1, v.y * w);
    atomicAdd(o + 2, v.z * w);
    atomicAdd(o + 3, v.w * w);
}

__global__ void elu_kernel(const float* __restrict__ in, float* __restrict__ out, size_t n) {
    size_t i = (size_t)blockIdx.x * blockDim.x + threadIdx.x;
    if (i < n) { float v = in[i]; out[i] = v > 0.f ? v : (__expf(v) - 1.f); }
}

__global__ void cls_kernel(const float* __restrict__ X, int Nn,
                           const float* __restrict__ W, const float* __restrict__ b,
                           float* __restrict__ out) {
    int n = blockIdx.x * blockDim.x + threadIdx.x;
    if (n >= Nn) return;
    const float* x = X + (size_t)n * HC;
    float s0 = b[0], s1 = b[1];
    for (int k = 0; k < HC; ++k) { float xv = x[k]; s0 += xv * W[2*k]; s1 += xv * W[2*k+1]; }
    out[2*n] = s0; out[2*n+1] = s1;
}

// ---------------------------------------------------------------------------
extern "C" void kernel_launch(void* const* d_in, const int* in_sizes, int n_in,
                              void* d_out, int out_size, void* d_ws, size_t ws_size,
                              hipStream_t stream) {
    const float* x_wallet = (const float*)d_in[0];
    const float* x_token  = (const float*)d_in[1];
    const float* lin_w_W  = (const float*)d_in[2];
    const float* lin_w_b  = (const float*)d_in[3];
    const float* lin_t_W  = (const float*)d_in[4];
    const float* lin_t_b  = (const float*)d_in[5];
    const float* Wsrc0    = (const float*)d_in[6];
    const float* Wdst0    = (const float*)d_in[7];
    const float* atts0    = (const float*)d_in[8];
    const float* attd0    = (const float*)d_in[9];
    const float* b0       = (const float*)d_in[10];
    const float* Wsrc1    = (const float*)d_in[11];
    const float* Wdst1    = (const float*)d_in[12];
    const float* atts1    = (const float*)d_in[13];
    const float* attd1    = (const float*)d_in[14];
    const float* b1       = (const float*)d_in[15];
    const float* cls_w_W  = (const float*)d_in[16];
    const float* cls_w_b  = (const float*)d_in[17];
    const float* cls_t_W  = (const float*)d_in[18];
    const float* cls_t_b  = (const float*)d_in[19];
    const int*   e_wt     = (const int*)d_in[20];
    const int*   e_tw     = (const int*)d_in[21];
    const int*   e_ww     = (const int*)d_in[22];

    float* ws = (float*)d_ws;
    float* xw    = ws; ws += (size_t)NWAL * HC;
    float* xt    = ws; ws += (size_t)NTOK * HC;
    float* xw2   = ws; ws += (size_t)NWAL * HC;
    float* xt2   = ws; ws += (size_t)NTOK * HC;
    float* hs    = ws; ws += (size_t)NWAL * HC;
    float* a_s   = ws; ws += (size_t)NWAL * 2;
    float* a_d   = ws; ws += (size_t)NWAL * 2;
    float* mbuf  = ws; ws += (size_t)NWAL * 2;
    float* den   = ws; ws += (size_t)NWAL * 2;
    float* exb   = ws; ws += (size_t)NEDG * 2;
    float* sfold = ws; ws += HC * 2;
    float* dfold = ws; ws += HC * 2;
    bf16_t* packw = (bf16_t*)ws; ws += (HC * HC) / 2;   // 128*128 bf16 = 8192 floats

    auto gemm = [&](const float* X, int M, int K, const float* W, int N,
                    float* Y, const float* bias) {
        pack_w<<<(K * N + 255) / 256, 256, 0, stream>>>(W, packw, K, N);
        int tiles = ((M + 15) >> 4) * (N >> 6);
        gemm_wmma_bf16<<<(tiles + 7) / 8, 256, 0, stream>>>(X, M, K, packw, N, Y, bias);
    };

    // input projections (f32 -> bf16 WMMA)
    gemm(x_wallet, NWAL, DW, lin_w_W, HID, xw, lin_w_b);
    gemm(x_token,  NTOK, DT, lin_t_W, HID, xt, lin_t_b);

    const int EB = (NEDG + 255) / 256;

    for (int L = 0; L < 2; ++L) {
        const float* Ws = L ? Wsrc1 : Wsrc0;
        const float* Wd = L ? Wdst1 : Wdst0;
        const float* As = L ? atts1 : atts0;
        const float* Ad = L ? attd1 : attd0;
        const float* bb = L ? b1 : b0;
        const int K = L ? HC : HID;
        const size_t WS = (size_t)K * HC;     // per-edge-type weight stride
        const size_t AS = 2 * HID;            // per-edge-type att stride

        // accumulators start at the (summed) biases
        bias_init<<<(int)(((size_t)NTOK*HC + 255)/256), 256, 0, stream>>>(xt2, (size_t)NTOK*HC, bb + 0*HC, nullptr);
        bias_init<<<(int)(((size_t)NWAL*HC + 255)/256), 256, 0, stream>>>(xw2, (size_t)NWAL*HC, bb + 1*HC, bb + 2*HC);

        auto conv = [&](const float* xs, int Ns, const float* xd, int Nd, int t,
                        const int* edges, float* outb) {
            fold_kernel<<<1, 256, 0, stream>>>(Ws + t*WS, As + t*AS, sfold, K);
            fold_kernel<<<1, 256, 0, stream>>>(Wd + t*WS, Ad + t*AS, dfold, K);
            gemm(xs, Ns, K, Ws + t*WS, HC, hs, nullptr);                       // hs = xs @ Wsrc
            scores_kernel<<<(Ns + 255)/256, 256, 0, stream>>>(xs, Ns, K, sfold, a_s);
            scores_kernel<<<(Nd + 255)/256, 256, 0, stream>>>(xd, Nd, K, dfold, a_d);
            init_md<<<(Nd*2 + 255)/256, 256, 0, stream>>>(mbuf, den, Nd*2);
            edge_pass1<<<EB, 256, 0, stream>>>(edges, edges + NEDG, a_s, a_d, exb, mbuf, NEDG);
            edge_pass2<<<EB, 256, 0, stream>>>(edges + NEDG, exb, mbuf, den, NEDG);
            edge_pass3<<<(NEDG + 7)/8, 256, 0, stream>>>(edges, edges + NEDG, exb, den, hs, outb, NEDG);
        };

        conv(xw, NWAL, xt, NTOK, 0, e_wt, xt2);   // wallet -> token
        conv(xt, NTOK, xw, NWAL, 1, e_tw, xw2);   // token  -> wallet
        conv(xw, NWAL, xw, NWAL, 2, e_ww, xw2);   // wallet -> wallet

        elu_kernel<<<(int)(((size_t)NWAL*HC + 255)/256), 256, 0, stream>>>(xw2, xw, (size_t)NWAL*HC);
        elu_kernel<<<(int)(((size_t)NTOK*HC + 255)/256), 256, 0, stream>>>(xt2, xt, (size_t)NTOK*HC);
    }

    float* out = (float*)d_out;
    cls_kernel<<<(NWAL + 255)/256, 256, 0, stream>>>(xw, NWAL, cls_w_W, cls_w_b, out);
    cls_kernel<<<(NTOK + 255)/256, 256, 0, stream>>>(xt, NTOK, cls_t_W, cls_t_b, out + (size_t)NWAL*2);
}
